// BaseEngine_20435454394998
// MI455X (gfx1250) — matrix operand order repaired and checked by reading, not compiled, not run on hardware
//
#include <hip/hip_runtime.h>
#include <hip/hip_bf16.h>
#include <math.h>

typedef __attribute__((ext_vector_type(16))) __bf16 v16bf;
typedef __attribute__((ext_vector_type(2)))  __bf16 v2bf;
typedef __attribute__((ext_vector_type(8)))  float  v8f;

#define NCELLS 131072
#define HIDD   128

// ---- workspace byte offsets ----
static const size_t OFF_PW1A  = 0;        // 128x128 bf16 packed B-frags
static const size_t OFF_PW1G  = 32768;
static const size_t OFF_PW2A  = 65536;    // 64x128
static const size_t OFF_PW2G  = 81920;    // negated
static const size_t OFF_PWIH  = 98304;    // 384x64
static const size_t OFF_PWHH  = 147456;   // 384x128
static const size_t OFF_VEC   = 245760;   // 1472 floats
static const size_t OFF_PESUM = 262144;   // 1024 f
static const size_t OFF_PTSUM = 266240;   // 1024 f
static const size_t OFF_PCOMB = 270336;   // 1024*64 f
static const size_t OFF_PFSUM = 532480;   // 1024*128 f
static const size_t OFF_FMEAN = 1056768;  // 8*128 f
static const size_t OFF_GO    = 1060864;  // 128 f

// vec float sub-offsets
#define V_XPA  0
#define V_XPG  128
#define V_B2   256
#define V_BIH  320
#define V_BHH  704
#define V_WCOL 1088

// one-time packing path: explicit RNE (matches hardware rounding)
__device__ __forceinline__ unsigned int bf16r(float x){
  unsigned int u = __float_as_uint(x);
  return (u + 0x7FFFu + ((u >> 16) & 1u)) >> 16;
}

// hot path: native fptrunc -> v_cvt_*bf16* on gfx1250
__device__ __forceinline__ unsigned int pk2(float lo, float hi){
  union { v2bf v; unsigned int u; } t;
  t.v[0] = (__bf16)lo; t.v[1] = (__bf16)hi;
  return t.u;
}

union BFrag {
  unsigned int u[8];
  uint4 q[2];
  v16bf v;
};

__device__ __forceinline__ v8f wmma_bf16(v16bf a, v16bf b, v8f c){
  return __builtin_amdgcn_wmma_f32_16x16x32_bf16(false, a, false, b, (short)0, c, false, false);
}

__device__ __forceinline__ BFrag load_bfrag(const unsigned short* base, int frag, int lane){
  BFrag r;
  const uint4* p = (const uint4*)(base + ((size_t)frag * 32 + lane) * 16);
  r.q[0] = p[0]; r.q[1] = p[1];
  return r;
}

__device__ __forceinline__ float fast_sigmoid(float x){
  return 1.f / (1.f + __expf(-x));
}
__device__ __forceinline__ float fast_tanh(float x){
  float e = __expf(2.f * x);
  return 1.f - 2.f / (e + 1.f);
}

// ---- pack a weight matrix [Nout][ldw] (cols col0..col0+Kin) into WMMA-B fragments ----
// B[k][n] = W[n][k];  frag(kt,nt): lane -> n = nt*16 + (lane&15), elem e -> k = kt*32 + (e>>3)*16 + (lane>>4)*8 + (e&7)
__global__ void pack_b(const float* __restrict__ W, unsigned short* __restrict__ dst,
                       int Nout, int Kin, int ldw, int col0, float scale){
  int idx = blockIdx.x * 256 + threadIdx.x;
  int total = Nout * Kin;
  if (idx >= total) return;
  int NT = Nout >> 4;
  int e    = idx & 15;
  int lane = (idx >> 4) & 31;
  int f    = idx >> 9;
  int kt = f / NT, nt = f - kt * NT;
  int half = lane >> 4;
  int k = kt * 32 + (e >> 3) * 16 + half * 8 + (e & 7);
  int n = nt * 16 + (lane & 15);
  float v = W[(size_t)n * ldw + col0 + k] * scale;
  dst[idx] = (unsigned short)bf16r(v);
}

// ---- precompute folded vectors ----
__global__ void prep_vec(const float* __restrict__ x,
                         const float* __restrict__ W1a, const float* __restrict__ b1a,
                         const float* __restrict__ W1g, const float* __restrict__ b1g,
                         const float* __restrict__ b2a, const float* __restrict__ b2g,
                         const float* __restrict__ Wih, const float* __restrict__ bih,
                         const float* __restrict__ bhh, float* __restrict__ vec){
  int i = blockIdx.x * 256 + threadIdx.x;
  if (i < 128){
    float v = b1a[i];
    for (int q = 0; q < 64; ++q) v += x[q] * W1a[(size_t)i * 192 + q];
    vec[V_XPA + i] = v;
  } else if (i < 256){
    int j = i - 128;
    float v = b1g[j];
    for (int q = 0; q < 64; ++q) v += x[q] * W1g[(size_t)j * 192 + q];
    vec[V_XPG + j] = v;
  } else if (i < 320){
    int j = i - 256; vec[V_B2 + j] = b2a[j] - b2g[j];
  } else if (i < 704){
    int j = i - 320; vec[V_BIH + j] = bih[j];
  } else if (i < 1088){
    int j = i - 704; vec[V_BHH + j] = bhh[j];
  } else if (i < 1472){
    int j = i - 1088; vec[V_WCOL + j] = Wih[(size_t)j * 65 + 64];
  }
}

// ---- main per-cell kernel: 256 threads = 8 waves, 16 cells/wave, 128 cells/block ----
__global__ __launch_bounds__(256) void cells_kernel(const float* __restrict__ hiddens,
                                                    unsigned char* __restrict__ ws_b,
                                                    float* __restrict__ newh_out){
  const unsigned short* PW1A = (const unsigned short*)(ws_b + OFF_PW1A);
  const unsigned short* PW1G = (const unsigned short*)(ws_b + OFF_PW1G);
  const unsigned short* PW2A = (const unsigned short*)(ws_b + OFF_PW2A);
  const unsigned short* PW2G = (const unsigned short*)(ws_b + OFF_PW2G);
  const unsigned short* PWIH = (const unsigned short*)(ws_b + OFF_PWIH);
  const unsigned short* PWHH = (const unsigned short*)(ws_b + OFF_PWHH);
  const float* vec = (const float*)(ws_b + OFF_VEC);
  float* part_esum = (float*)(ws_b + OFF_PESUM);
  float* part_tsum = (float*)(ws_b + OFF_PTSUM);
  float* part_comb = (float*)(ws_b + OFF_PCOMB);
  float* part_fsum = (float*)(ws_b + OFF_PFSUM);

  __shared__ __bf16 stage[8][16][128];   // per-wave C->A re-fragment staging
  __shared__ float s_comb[8][64];
  __shared__ float s_fsum[8][128];
  __shared__ float s_sc[8][2];

  const int tid  = threadIdx.x;
  const int w    = tid >> 5;
  const int lane = tid & 31;
  const int half = lane >> 4;
  const int l15  = lane & 15;
  const int cell0 = (blockIdx.x * 8 + w) * 16;

  // --- A fragments of h (16 cells x K=128) ---
  BFrag Ah[4];
  {
    const float* hrow = hiddens + (size_t)(cell0 + l15) * HIDD;
#pragma unroll
    for (int kt = 0; kt < 4; ++kt){
      int b0 = kt * 32 + half * 8;
      int b1 = b0 + 16;
      float4 f0 = *(const float4*)(hrow + b0);
      float4 f1 = *(const float4*)(hrow + b0 + 4);
      float4 f2 = *(const float4*)(hrow + b1);
      float4 f3 = *(const float4*)(hrow + b1 + 4);
      Ah[kt].u[0] = pk2(f0.x, f0.y); Ah[kt].u[1] = pk2(f0.z, f0.w);
      Ah[kt].u[2] = pk2(f1.x, f1.y); Ah[kt].u[3] = pk2(f1.z, f1.w);
      Ah[kt].u[4] = pk2(f2.x, f2.y); Ah[kt].u[5] = pk2(f2.z, f2.w);
      Ah[kt].u[6] = pk2(f3.x, f3.y); Ah[kt].u[7] = pk2(f3.z, f3.w);
    }
  }

  const __bf16* srow = &stage[w][l15][0];

  // --- h1a = relu(h @ W1a'.T + xpa) ---
#pragma unroll
  for (int nt = 0; nt < 8; ++nt){
    v8f acc = {0,0,0,0,0,0,0,0};
#pragma unroll
    for (int kt = 0; kt < 4; ++kt)
      acc = wmma_bf16(Ah[kt].v, load_bfrag(PW1A, kt*8 + nt, lane).v, acc);
    float xp = vec[V_XPA + nt*16 + l15];
#pragma unroll
    for (int r = 0; r < 8; ++r){
      float vv = acc[r] + xp; vv = vv > 0.f ? vv : 0.f;
      stage[w][r + half*8][nt*16 + l15] = (__bf16)vv;
    }
  }
  BFrag A1a[4];
#pragma unroll
  for (int kt = 0; kt < 4; ++kt){
    A1a[kt].q[0] = *(const uint4*)(srow + kt*32 + half*8);
    A1a[kt].q[1] = *(const uint4*)(srow + kt*32 + 16 + half*8);
  }

  // --- h1g = relu(h @ W1g'.T + xpg) ---
#pragma unroll
  for (int nt = 0; nt < 8; ++nt){
    v8f acc = {0,0,0,0,0,0,0,0};
#pragma unroll
    for (int kt = 0; kt < 4; ++kt)
      acc = wmma_bf16(Ah[kt].v, load_bfrag(PW1G, kt*8 + nt, lane).v, acc);
    float xp = vec[V_XPG + nt*16 + l15];
#pragma unroll
    for (int r = 0; r < 8; ++r){
      float vv = acc[r] + xp; vv = vv > 0.f ? vv : 0.f;
      stage[w][r + half*8][nt*16 + l15] = (__bf16)vv;
    }
  }
  BFrag A1g[4];
#pragma unroll
  for (int kt = 0; kt < 4; ++kt){
    A1g[kt].q[0] = *(const uint4*)(srow + kt*32 + half*8);
    A1g[kt].q[1] = *(const uint4*)(srow + kt*32 + 16 + half*8);
  }

  // --- out = h1a @ W2a.T - h1g @ W2g.T + (b2a-b2g)   (W2g packed negated) ---
  v8f outacc[4];
#pragma unroll
  for (int nt = 0; nt < 4; ++nt){
    v8f acc = {0,0,0,0,0,0,0,0};
#pragma unroll
    for (int kt = 0; kt < 4; ++kt)
      acc = wmma_bf16(A1a[kt].v, load_bfrag(PW2A, kt*4 + nt, lane).v, acc);
#pragma unroll
    for (int kt = 0; kt < 4; ++kt)
      acc = wmma_bf16(A1g[kt].v, load_bfrag(PW2G, kt*4 + nt, lane).v, acc);
    float bb = vec[V_B2 + nt*16 + l15];
#pragma unroll
    for (int r = 0; r < 8; ++r) acc[r] += bb;
    outacc[nt] = acc;
  }

  // --- t = mean(out^2) per cell (cell m = r + half*8), e = exp(t) ---
  float tcell[8], ecell[8];
#pragma unroll
  for (int r = 0; r < 8; ++r){
    float sq = 0.f;
#pragma unroll
    for (int nt = 0; nt < 4; ++nt){ float o = outacc[nt][r]; sq += o * o; }
#pragma unroll
    for (int mk = 1; mk < 16; mk <<= 1) sq += __shfl_xor(sq, mk, 32);
    tcell[r] = sq * (1.0f / 64.0f);
    ecell[r] = __expf(tcell[r]);
  }

  // --- wave partials: sum(e), sum(t), sum(e*out) ---
  {
    float tp = 0.f, ep = 0.f;
#pragma unroll
    for (int r = 0; r < 8; ++r){ tp += tcell[r]; ep += ecell[r]; }
    tp += __shfl_xor(tp, 16, 32);
    ep += __shfl_xor(ep, 16, 32);
    if (lane == 0){ s_sc[w][0] = ep; s_sc[w][1] = tp; }
#pragma unroll
    for (int nt = 0; nt < 4; ++nt){
      float cp = 0.f;
#pragma unroll
      for (int r = 0; r < 8; ++r) cp += ecell[r] * outacc[nt][r];
      cp += __shfl_xor(cp, 16, 32);
      if (lane < 16) s_comb[w][nt*16 + lane] = cp;
    }
  }

  // --- re-fragment out (K=64) as A for the GRU gi GEMM ---
#pragma unroll
  for (int nt = 0; nt < 4; ++nt)
#pragma unroll
    for (int r = 0; r < 8; ++r)
      stage[w][r + half*8][nt*16 + l15] = (__bf16)outacc[nt][r];
  BFrag Ao[2];
#pragma unroll
  for (int kt = 0; kt < 2; ++kt){
    Ao[kt].q[0] = *(const uint4*)(srow + kt*32 + half*8);
    Ao[kt].q[1] = *(const uint4*)(srow + kt*32 + 16 + half*8);
  }

  // --- GRU: per 16-col tile compute r/z/n gate GEMMs and elementwise update ---
  const float* bihv = vec + V_BIH;
  const float* bhhv = vec + V_BHH;
  const float* wcol = vec + V_WCOL;
  for (int nt = 0; nt < 8; ++nt){
    v8f gr = {0,0,0,0,0,0,0,0}, gz = {0,0,0,0,0,0,0,0}, gn = {0,0,0,0,0,0,0,0};
    v8f hr = {0,0,0,0,0,0,0,0}, hz = {0,0,0,0,0,0,0,0}, hn = {0,0,0,0,0,0,0,0};
#pragma unroll
    for (int kt = 0; kt < 2; ++kt){
      gr = wmma_bf16(Ao[kt].v, load_bfrag(PWIH, kt*24 + nt,      lane).v, gr);
      gz = wmma_bf16(Ao[kt].v, load_bfrag(PWIH, kt*24 + 8 + nt,  lane).v, gz);
      gn = wmma_bf16(Ao[kt].v, load_bfrag(PWIH, kt*24 + 16 + nt, lane).v, gn);
    }
#pragma unroll
    for (int kt = 0; kt < 4; ++kt){
      hr = wmma_bf16(Ah[kt].v, load_bfrag(PWHH, kt*24 + nt,      lane).v, hr);
      hz = wmma_bf16(Ah[kt].v, load_bfrag(PWHH, kt*24 + 8 + nt,  lane).v, hz);
      hn = wmma_bf16(Ah[kt].v, load_bfrag(PWHH, kt*24 + 16 + nt, lane).v, hn);
    }
    int jj = nt*16 + l15;
    float bir = bihv[jj], biz = bihv[jj+128], bin = bihv[jj+256];
    float bhr = bhhv[jj], bhz = bhhv[jj+128], bhn = bhhv[jj+256];
    float wcr = wcol[jj], wcz = wcol[jj+128], wcn = wcol[jj+256];
    float fp = 0.f;
#pragma unroll
    for (int r = 0; r < 8; ++r){
      float t  = tcell[r];
      float gi_r = gr[r] + bir + t * wcr;
      float gi_z = gz[r] + biz + t * wcz;
      float gi_n = gn[r] + bin + t * wcn;
      float gh_r = hr[r] + bhr;
      float gh_z = hz[r] + bhz;
      float gh_n = hn[r] + bhn;
      float rg = fast_sigmoid(gi_r + gh_r);
      float zg = fast_sigmoid(gi_z + gh_z);
      float ng = fast_tanh(gi_n + rg * gh_n);
      size_t cidx = (size_t)(cell0 + r + half*8) * HIDD + jj;
      float hold = hiddens[cidx];
      float nh = (1.f - zg) * ng + zg * hold;
      newh_out[cidx] = nh;
      fp += nh;
    }
    fp += __shfl_xor(fp, 16, 32);
    if (lane < 16) s_fsum[w][nt*16 + lane] = fp;
  }

  __syncthreads();
  // --- deterministic block reduction of wave partials ---
  if (tid < 64){
    float v = 0.f;
#pragma unroll
    for (int ww = 0; ww < 8; ++ww) v += s_comb[ww][tid];
    part_comb[(size_t)blockIdx.x * 64 + tid] = v;
  } else if (tid < 192){
    int j = tid - 64;
    float v = 0.f;
#pragma unroll
    for (int ww = 0; ww < 8; ++ww) v += s_fsum[ww][j];
    part_fsum[(size_t)blockIdx.x * 128 + j] = v;
  } else if (tid == 192){
    float v = 0.f;
    for (int ww = 0; ww < 8; ++ww) v += s_sc[ww][0];
    part_esum[blockIdx.x] = v;
  } else if (tid == 193){
    float v = 0.f;
    for (int ww = 0; ww < 8; ++ww) v += s_sc[ww][1];
    part_tsum[blockIdx.x] = v;
  }
}

// ---- final reduction: softmax combine -> pred, t.mean, faction means, global opinion ----
__global__ void reduce_kernel(unsigned char* __restrict__ ws_b,
                              const float* __restrict__ Wo, const float* __restrict__ bo,
                              float* __restrict__ out){
  const float* part_esum = (const float*)(ws_b + OFF_PESUM);
  const float* part_tsum = (const float*)(ws_b + OFF_PTSUM);
  const float* part_comb = (const float*)(ws_b + OFF_PCOMB);
  const float* part_fsum = (const float*)(ws_b + OFF_PFSUM);
  float* fmean = (float*)(ws_b + OFF_FMEAN);
  float* go    = (float*)(ws_b + OFF_GO);

  __shared__ float s_comb[64];
  __shared__ float s_esum, s_tsum;
  int tid = threadIdx.x;

  if (tid < 64){
    float v = 0.f;
    for (int b = 0; b < 1024; ++b) v += part_comb[(size_t)b * 64 + tid];
    s_comb[tid] = v;
  } else if (tid == 64){
    float v = 0.f;
    for (int b = 0; b < 1024; ++b) v += part_esum[b];
    s_esum = v;
  } else if (tid == 65){
    float v = 0.f;
    for (int b = 0; b < 1024; ++b) v += part_tsum[b];
    s_tsum = v;
  }
  // faction means: 8 factions x 128 dims; each faction = 128 consecutive blocks
  for (int i = tid; i < 1024; i += 256){
    int f = i >> 7, j = i & 127;
    float v = 0.f;
    for (int b = 0; b < 128; ++b) v += part_fsum[(size_t)(f * 128 + b) * 128 + j];
    fmean[i] = v * (1.0f / 16384.0f);
  }
  __syncthreads();
  if (tid < 128){
    float v = 0.f;
    for (int f = 0; f < 8; ++f) v += fmean[f * 128 + tid];
    go[tid] = v * (1.0f / 8.0f);
  }
  if (tid < 64){
    float inv = 1.0f / s_esum;
    float v = 0.f;
    for (int n = 0; n < 64; ++n) v += s_comb[n] * inv * Wo[(size_t)tid * 64 + n];
    out[tid] = v + bo[tid];
  }
  if (tid == 64) out[64] = s_tsum * (1.0f / (float)NCELLS);
}

// ---- in-place faction sync on new_h ----
__global__ void sync_kernel(float* __restrict__ newh,
                            const float* __restrict__ fmean, const float* __restrict__ go,
                            const int* __restrict__ stepp){
  unsigned int idx = blockIdx.x * 256u + threadIdx.x;   // 131072*128 elements
  unsigned int c = idx >> 7, j = idx & 127u;
  unsigned int f = c >> 14, pos = c & 16383u;
  float v = newh[idx];
  v = 0.85f * v + 0.15f * fmean[f * 128 + j];
  if (*stepp > 5 && pos < 4096u) v = 0.85f * v + 0.15f * go[j];
  newh[idx] = v;
}

extern "C" void kernel_launch(void* const* d_in, const int* in_sizes, int n_in,
                              void* d_out, int out_size, void* d_ws, size_t ws_size,
                              hipStream_t stream){
  const float* x    = (const float*)d_in[0];
  const float* hid  = (const float*)d_in[1];
  const float* W1a  = (const float*)d_in[2];
  const float* b1a  = (const float*)d_in[3];
  const float* W2a  = (const float*)d_in[4];
  const float* b2a  = (const float*)d_in[5];
  const float* W1g  = (const float*)d_in[6];
  const float* b1g  = (const float*)d_in[7];
  const float* W2g  = (const float*)d_in[8];
  const float* b2g  = (const float*)d_in[9];
  const float* Wih  = (const float*)d_in[10];
  const float* Whh  = (const float*)d_in[11];
  const float* bih  = (const float*)d_in[12];
  const float* bhh  = (const float*)d_in[13];
  const float* Wo   = (const float*)d_in[14];
  const float* bo   = (const float*)d_in[15];
  const int*   step = (const int*)d_in[16];

  unsigned char* ws = (unsigned char*)d_ws;
  float* out = (float*)d_out;

  pack_b<<<64,  256, 0, stream>>>(W1a, (unsigned short*)(ws + OFF_PW1A), 128, 128, 192, 64,  1.f);
  pack_b<<<64,  256, 0, stream>>>(W1g, (unsigned short*)(ws + OFF_PW1G), 128, 128, 192, 64,  1.f);
  pack_b<<<32,  256, 0, stream>>>(W2a, (unsigned short*)(ws + OFF_PW2A),  64, 128, 128,  0,  1.f);
  pack_b<<<32,  256, 0, stream>>>(W2g, (unsigned short*)(ws + OFF_PW2G),  64, 128, 128,  0, -1.f);
  pack_b<<<96,  256, 0, stream>>>(Wih, (unsigned short*)(ws + OFF_PWIH), 384,  64,  65,  0,  1.f);
  pack_b<<<192, 256, 0, stream>>>(Whh, (unsigned short*)(ws + OFF_PWHH), 384, 128, 128,  0,  1.f);
  prep_vec<<<6, 256, 0, stream>>>(x, W1a, b1a, W1g, b1g, b2a, b2g, Wih, bih, bhh,
                                  (float*)(ws + OFF_VEC));
  cells_kernel<<<1024, 256, 0, stream>>>(hid, ws, out + 65);
  reduce_kernel<<<1, 256, 0, stream>>>(ws, Wo, bo, out);
  sync_kernel<<<65536, 256, 0, stream>>>(out + 65,
                                         (const float*)(ws + OFF_FMEAN),
                                         (const float*)(ws + OFF_GO), step);
}
